// CombinedLoss_55052890800433
// MI455X (gfx1250) — compile-verified, matrix operand order checked by reading, not run on hardware
//
#include <hip/hip_runtime.h>
#include <hip/hip_bf16.h>

// ---------------------------------------------------------------------------
// CombinedLoss (YOLO DIoU + NMS + confidence penalty) for MI455X / gfx1250.
// B=128, H=W=64, A=9 -> N=36864 anchors per image, K=T=100.
// Memory-bound: one 94MB streaming pass for top-k (~4us HBM floor), tiny tail
// kernels, WMMA (f16 hi/lo split, f32 accumulate) for the final reduction.
// ---------------------------------------------------------------------------

#define BB   128
#define HH   64
#define WW   64
#define AA   9
#define NN   (HH * WW * AA)      // 36864 = 36 * 1024
#define KK   100
#define DSZ  1024.0f
#define STR  64.0f
#define EPSV 1e-7f
#define PLANE 256                // padded reduction plane (128 data + 128 zeros)

typedef __attribute__((ext_vector_type(16))) _Float16 v16h;
typedef __attribute__((ext_vector_type(8)))  float    v8f;

__device__ __forceinline__ float sigmoidf_(float x) {
    return 1.0f / (1.0f + expf(-x));
}

__device__ __forceinline__ float box_iou_(const float* a, const float* b) {
    float aa = (a[2] - a[0]) * (a[3] - a[1]);
    float ab = (b[2] - b[0]) * (b[3] - b[1]);
    float lx = fmaxf(a[0], b[0]);
    float ly = fmaxf(a[1], b[1]);
    float rx = fminf(a[2], b[2]);
    float ry = fminf(a[3], b[3]);
    float w  = fmaxf(rx - lx, 0.0f);
    float h  = fmaxf(ry - ly, 0.0f);
    float inter = w * h;
    return inter / (aa + ab - inter + EPSV);
}

__device__ __forceinline__ void clip_fix_(float c0, float c1, float c2, float c3, float* o) {
    c0 = fminf(fmaxf(c0, 0.0f), DSZ);
    c1 = fminf(fmaxf(c1, 0.0f), DSZ);
    c2 = fminf(fmaxf(c2, 0.0f), DSZ);
    c3 = fminf(fmaxf(c3, 0.0f), DSZ);
    float x1 = fminf(c0, c2), y1 = fminf(c1, c3);
    float x2 = fmaxf(c0, c2), y2 = fmaxf(c1, c3);
    if (x1 == x2) x2 = x1 + 1.0f;
    if (y1 == y2) y2 = y1 + 1.0f;
    o[0] = x1; o[1] = y1; o[2] = x2; o[3] = y2;
}

// key = ordered conf bits (conf>0 so raw bits are monotonic) << 32 | ~index
// -> max key == highest conf, ties broken toward smaller index (jax top_k order)
__device__ __forceinline__ unsigned long long mk_key_(float conf, int n) {
    unsigned long long hi = (unsigned long long)__float_as_uint(conf);
    return (hi << 32) | (unsigned long long)(0xFFFFFFFFu - (unsigned)n);
}

// ---------------------------------------------------------------------------
// Kernel 1: exact top-100 confidences per image (one block per image).
// Thread t owns elements n = e*1024 + t, e in [0,36) (coalesced stream).
// ---------------------------------------------------------------------------
__global__ __launch_bounds__(1024) void topk_select_kernel(
    const float* __restrict__ out, int* __restrict__ sel_idx, float* __restrict__ sel_val)
{
    const int b   = blockIdx.x;
    const int tid = threadIdx.x;
    const int lane = tid & 31;
    const int wv   = tid >> 5;

    __shared__ unsigned long long warr[32];
    __shared__ unsigned long long bestk_sm;

    const float* obase = out + (size_t)b * NN * 5;

    // Initial per-thread max over owned 36 elements (streaming, prefetch ahead).
    unsigned long long pk = 0ull;
    unsigned long long removed = 0ull;   // bit e set if element e*1024+tid was selected
    for (int e = 0; e < 36; ++e) {
        int n = e * 1024 + tid;
        __builtin_prefetch(obase + (size_t)(n + 4096) * 5 + 4, 0, 0);
        float conf = sigmoidf_(obase[(size_t)n * 5 + 4]);
        unsigned long long key = mk_key_(conf, n);
        if (key > pk) pk = key;
    }

    for (int it = 0; it < KK; ++it) {
        // wave32 max reduce
        unsigned long long k = pk;
        #pragma unroll
        for (int off = 16; off > 0; off >>= 1) {
            unsigned long long o = __shfl_xor(k, off, 32);
            if (o > k) k = o;
        }
        if (lane == 0) warr[wv] = k;
        __syncthreads();
        if (tid < 32) {
            unsigned long long k2 = warr[tid];
            #pragma unroll
            for (int off = 16; off > 0; off >>= 1) {
                unsigned long long o = __shfl_xor(k2, off, 32);
                if (o > k2) k2 = o;
            }
            if (tid == 0) bestk_sm = k2;
        }
        __syncthreads();
        unsigned long long bk = bestk_sm;
        int n = (int)(0xFFFFFFFFu - (unsigned)(bk & 0xFFFFFFFFull));
        if (tid == 0) {
            sel_idx[b * KK + it] = n;
            sel_val[b * KK + it] = __uint_as_float((unsigned)(bk >> 32));
        }
        int owner = n & 1023;
        if (tid == owner) {
            removed |= (1ull << (n >> 10));
            unsigned long long np = 0ull;
            for (int e = 0; e < 36; ++e) {
                if (removed & (1ull << e)) continue;
                int m = e * 1024 + tid;
                float conf = sigmoidf_(obase[(size_t)m * 5 + 4]);
                unsigned long long key = mk_key_(conf, m);
                if (key > np) np = key;
            }
            pk = np;
        }
        __syncthreads();
    }
}

// ---------------------------------------------------------------------------
// Kernel 2: per-image decode + clip/fix + greedy NMS + DIoU + penalty sums.
// One block (128 threads) per image. res layout: res[q*PLANE + b], q=0..3
//   q0 = sum_t diou_loss, q1 = sum low-pen, q2 = sum high-pen, q3 = kept count
// Upper half of each plane (indices 128..255) is zeroed so the WMMA kernel
// can load fragments branchlessly.
// ---------------------------------------------------------------------------
__global__ __launch_bounds__(128) void per_image_kernel(
    const float* __restrict__ out, const float* __restrict__ tboxes,
    const float* __restrict__ anchors, const int* __restrict__ sel_idx,
    const float* __restrict__ sel_val, float* __restrict__ res)
{
    const int b = blockIdx.x;
    const int t = threadIdx.x;

    __shared__ float bx[KK][4];
    __shared__ float tb[KK][4];
    __shared__ float vals[KK];
    __shared__ int   keep[KK];
    __shared__ float keepf[KK];
    __shared__ float anc[AA * 2];
    __shared__ float red[128];

    if (t < AA * 2) anc[t] = anchors[t];
    __syncthreads();

    if (t < KK) {
        int   n = sel_idx[b * KK + t];
        float v = sel_val[b * KK + t];
        vals[t] = v;
        keep[t] = (v >= 0.5f) ? 1 : 0;

        int a = n % AA;
        int w = (n / AA) % WW;
        int h = n / (AA * WW);
        const float* o = out + ((size_t)b * NN + n) * 5;
        float px = (sigmoidf_(o[0]) + (float)w) * STR;
        float py = (sigmoidf_(o[1]) + (float)h) * STR;
        float pw = expf(o[2]) * anc[a * 2 + 0] * DSZ;
        float ph = expf(o[3]) * anc[a * 2 + 1] * DSZ;
        clip_fix_(px - pw * 0.5f, py - ph * 0.5f, px + pw * 0.5f, py + ph * 0.5f, bx[t]);

        const float* tp = tboxes + ((size_t)b * KK + t) * 4;
        clip_fix_(tp[0], tp[1], tp[2], tp[3], tb[t]);
    }
    __syncthreads();

    // Greedy NMS, identical to the reference fori_loop (keep starts at cmask).
    for (int i = 0; i < KK; ++i) {
        int ki = keep[i];
        if (ki && t < KK && t > i && keep[t]) {
            if (box_iou_(bx[i], bx[t]) > 0.5f) keep[t] = 0;
        }
        __syncthreads();
    }
    if (t < KK) keepf[t] = (float)keep[t];
    __syncthreads();

    // DIoU per target (first-occurrence argmax over preds, suppressed preds = 0).
    float myd = 0.0f;
    if (t < KK) {
        float bv = -1.0f;
        int   bi = 0;
        for (int i = 0; i < KK; ++i) {
            float iou = box_iou_(bx[i], tb[t]) * keepf[i];
            if (iou > bv) { bv = iou; bi = i; }
        }
        float mx = fmaxf(bv, 0.0f);
        const float* pb = bx[bi];
        float pcx = (pb[0] + pb[2]) * 0.5f, pcy = (pb[1] + pb[3]) * 0.5f;
        float tcx = (tb[t][0] + tb[t][2]) * 0.5f, tcy = (tb[t][1] + tb[t][3]) * 0.5f;
        float cd  = (pcx - tcx) * (pcx - tcx) + (pcy - tcy) * (pcy - tcy);
        float ex1 = fminf(pb[0], tb[t][0]), ey1 = fminf(pb[1], tb[t][1]);
        float ex2 = fmaxf(pb[2], tb[t][2]), ey2 = fmaxf(pb[3], tb[t][3]);
        float dg  = (ex2 - ex1) * (ex2 - ex1) + (ey2 - ey1) * (ey2 - ey1);
        myd = 1.0f - (mx - cd / (dg + EPSV));
    }
    float myl = 0.0f, myh = 0.0f, myc = 0.0f;
    if (t < KK) {
        float kf = keepf[t];
        myl = fmaxf(-vals[t], 0.0f) * kf;
        myh = fmaxf(vals[t] - 1.0f, 0.0f) * kf;
        myc = kf;
    }

    float q[4] = {myd, myl, myh, myc};
    #pragma unroll
    for (int qi = 0; qi < 4; ++qi) {
        red[t] = q[qi];
        __syncthreads();
        #pragma unroll
        for (int s = 64; s > 0; s >>= 1) {
            if (t < s) red[t] += red[t + s];
            __syncthreads();
        }
        if (t == 0) {
            res[qi * PLANE + b]       = red[0];   // data half
            res[qi * PLANE + 128 + b] = 0.0f;     // zero pad half (re-init every call)
        }
        __syncthreads();
    }
}

// ---------------------------------------------------------------------------
// Kernel 3: final scalar via 4x v_wmma_f32_16x16x32_f16 (one wave).
// B = all-ones  =>  sum over all rows of D (at N=0) == sum of all A elements,
// independent of the A fragment layout. Each f32 value x is placed as two f16
// slots (hi = f16(x), lo = f16(x - hi)) so the f32-accumulated sum keeps ~f32
// accuracy. 512 A slots cover the 256-entry padded plane (upper half zeros),
// making the fragment fill branchless.
// ---------------------------------------------------------------------------
__global__ __launch_bounds__(32) void final_reduce_kernel(
    const float* __restrict__ res, float* __restrict__ out_scalar)
{
    const int l = threadIdx.x; // 0..31, single wave32, EXEC all ones
    float tot[4];

    v16h ones;
    #pragma unroll
    for (int s = 0; s < 16; ++s) ones[s] = (_Float16)1.0f;

    #pragma unroll
    for (int qi = 0; qi < 4; ++qi) {
        const float* d = res + qi * PLANE;
        v16h a;
        #pragma unroll
        for (int s = 0; s < 16; ++s) {
            int g = l * 16 + s;                 // A-fragment slot 0..511
            float x = d[g >> 1];                // always in-bounds (padded plane)
            _Float16 hi = (_Float16)x;
            a[s] = (g & 1) ? (_Float16)(x - (float)hi) : hi;
        }
        v8f c = {};
        c = __builtin_amdgcn_wmma_f32_16x16x32_f16(
                /*neg_a=*/false, a, /*neg_b=*/false, ones,
                /*c_mod=*/(short)0, c, /*reuse_a=*/false, /*reuse_b=*/false);
        // lane 0: D[0..7][0], lane 16: D[8..15][0]; rows are replicated across N.
        float s8 = c[0] + c[1] + c[2] + c[3] + c[4] + c[5] + c[6] + c[7];
        s8 += __shfl_xor(s8, 16, 32);
        tot[qi] = s8;
    }

    if (l == 0) {
        float n_kept = fmaxf(tot[3], 1.0f);
        float result = tot[0] / (float)(BB * KK)
                     + tot[1] / n_kept
                     + 0.5f * tot[2] / n_kept;
        out_scalar[0] = result;
    }
}

// ---------------------------------------------------------------------------
extern "C" void kernel_launch(void* const* d_in, const int* in_sizes, int n_in,
                              void* d_out, int out_size, void* d_ws, size_t ws_size,
                              hipStream_t stream) {
    const float* out_t   = (const float*)d_in[0];   // [128,64,64,9,5]
    const float* tboxes  = (const float*)d_in[1];   // [128,100,4]
    const float* anchors = (const float*)d_in[2];   // [9,2]
    float* result = (float*)d_out;

    char* ws = (char*)d_ws;
    int*   sel_idx = (int*)ws;                                   // 128*100 ints
    float* sel_val = (float*)(ws + BB * KK * sizeof(int));       // 128*100 floats
    float* res     = (float*)(ws + 2 * BB * KK * sizeof(int));   // 4*256 floats (padded planes)

    topk_select_kernel<<<BB, 1024, 0, stream>>>(out_t, sel_idx, sel_val);
    per_image_kernel<<<BB, 128, 0, stream>>>(out_t, tboxes, anchors, sel_idx, sel_val, res);
    final_reduce_kernel<<<1, 32, 0, stream>>>(res, result);
}
